// GNNPolicy_64845416235556
// MI455X (gfx1250) — matrix operand order repaired and checked by reading, not compiled, not run on hardware
//
#include <hip/hip_runtime.h>
#include <math.h>

#define N_NODES 50000
#define N_EDGES 800000
#define F_HID   128

typedef _Float16 h16 __attribute__((ext_vector_type(16)));
typedef float    f8  __attribute__((ext_vector_type(8)));

// ---------------------------------------------------------------- utility
__global__ void k_zero_f32(float* __restrict__ p, int n) {
    int i = blockIdx.x * blockDim.x + threadIdx.x;
    if (i < n) p[i] = 0.f;
}

// deg[col] += w   (target-side degree, no self loops)
__global__ void k_deg(const int* __restrict__ col, const float* __restrict__ w,
                      float* __restrict__ deg) {
    int e = blockIdx.x * blockDim.x + threadIdx.x;
    if (e < N_EDGES) atomicAdd(&deg[col[e]], w[e]);
}

__global__ void k_rsqrt(float* __restrict__ deg) {
    int i = blockIdx.x * blockDim.x + threadIdx.x;
    if (i < N_NODES) {
        float d = deg[i];
        deg[i] = d > 0.f ? rsqrtf(d) : 0.f;
    }
}

__global__ void k_normw(const int* __restrict__ row, const int* __restrict__ col,
                        const float* __restrict__ w, const float* __restrict__ dis,
                        float* __restrict__ nw) {
    int e = blockIdx.x * blockDim.x + threadIdx.x;
    if (e < N_EDGES) nw[e] = dis[row[e]] * w[e] * dis[col[e]];
}

// Xcat: 50000 x 32 f32. cols [0,6) = x (hop 0), cols [6k,6k+6) = hop k, 30..31 zero.
__global__ void k_init_xcat(const float* __restrict__ x, float* __restrict__ Xcat) {
    int idx = blockIdx.x * blockDim.x + threadIdx.x;   // 50000*32
    int i = idx >> 5, c = idx & 31;
    Xcat[idx] = (c < 6) ? x[i * 6 + c] : 0.f;
}

// W1 [5][6][128] f32 -> W1s 32x128 f16 (rows 30,31 zero). flat idx match: r=k*6+i.
__global__ void k_prep_w1(const float* __restrict__ W1, _Float16* __restrict__ W1s) {
    int idx = blockIdx.x * blockDim.x + threadIdx.x;   // 32*128
    if (idx < 32 * 128) {
        int r = idx >> 7;
        W1s[idx] = (r < 30) ? (_Float16)W1[idx] : (_Float16)0.f;
    }
}

// W2 [5][128][6] f32 -> W2s [5][128][16] f16 (cols 6..15 zero).
__global__ void k_prep_w2(const float* __restrict__ W2, _Float16* __restrict__ W2s) {
    int idx = blockIdx.x * blockDim.x + threadIdx.x;   // 5*128*16
    if (idx < 5 * 128 * 16) {
        int k = idx >> 11, rem = idx & 2047, r = rem >> 4, n = rem & 15;
        W2s[idx] = (n < 6) ? (_Float16)W2[(k * 128 + r) * 6 + n] : (_Float16)0.f;
    }
}

// layer-1 hop: Xcat[:, dstOff..+6) += scatter(nw * Xcat[row, srcOff..+6))
__global__ void k_prop1(const int* __restrict__ row, const int* __restrict__ col,
                        const float* __restrict__ nw, float* __restrict__ Xcat,
                        int srcOff, int dstOff) {
    int idx = blockIdx.x * blockDim.x + threadIdx.x;   // N_EDGES*8
    int e = idx >> 3, f = idx & 7;
    if (f < 6) {
        float v = nw[e] * Xcat[row[e] * 32 + srcOff + f];
        atomicAdd(&Xcat[col[e] * 32 + dstOff + f], v);
    }
}

// layer-2 hop: dst[col,:] += nw * src[row,:]   (128 feats per edge)
__global__ void k_prop2(const int* __restrict__ row, const int* __restrict__ col,
                        const float* __restrict__ nw,
                        const float* __restrict__ src, float* __restrict__ dst) {
    int idx = blockIdx.x * blockDim.x + threadIdx.x;   // N_EDGES*128, exact grid
    int e = idx >> 7, f = idx & 127;
    atomicAdd(&dst[col[e] * 128 + f], nw[e] * src[row[e] * 128 + f]);
}

// ------------------------------------------------- WMMA GEMM 1: h = relu(Xcat @ W1s + b1)
// Xcat: M=50000 x K=32 f32 (converted), W1s: 32 x 128 f16, one wave per 16x16 tile.
__global__ void __launch_bounds__(32)
k_gemm1(const float* __restrict__ Xcat, const _Float16* __restrict__ W1s,
        const float* __restrict__ b1, float* __restrict__ h) {
    const int lane = threadIdx.x;
    const int hf   = lane >> 4;      // lane half-group
    const int mr   = lane & 15;
    const int tm   = blockIdx.y;     // 0..3124
    const int tn   = blockIdx.x;     // 0..7
    const int m    = tm * 16 + mr;
    const int n    = tn * 16 + mr;

    const float* ar = Xcat + m * 32;
    const int lowK = hf * 8;
    h16 a, b;
#pragma unroll
    for (int e = 0; e < 8; ++e) {                 // ISA 16-bit A layout
        a[e]     = (_Float16)ar[lowK + e];        // K = lowK + e
        a[8 + e] = (_Float16)ar[16 + lowK + e];   // K = 16 + lowK + e
    }
#pragma unroll
    for (int e = 0; e < 16; ++e)                  // B: row K = 16*(e/8) + lowK + e%8, col n
        b[e] = W1s[(((e >> 3) << 4) + lowK + (e & 7)) * 128 + n];

    f8 c = {};
    c = __builtin_amdgcn_wmma_f32_16x16x32_f16(false, a, false, b, (short)0, c,
                                               false, false);
    const float bias = b1[n];
#pragma unroll
    for (int r = 0; r < 8; ++r) {                 // D: row r + 8*hf, col n
        float v = c[r] + bias;
        h[(tm * 16 + r + 8 * hf) * 128 + n] = v > 0.f ? v : 0.f;
    }
}

// ------------------------------------------------- WMMA GEMM 2: out2 += src @ W2s_k
// src: 50000 x 128 f32, W2s_k: 128 x 16 f16, out2: 50000 x 16 f32 accumulator.
__global__ void __launch_bounds__(32)
k_gemm2(const float* __restrict__ src, const _Float16* __restrict__ W2sk,
        float* __restrict__ out2) {
    const int lane = threadIdx.x;
    const int hf   = lane >> 4;
    const int mr   = lane & 15;
    const int tm   = blockIdx.x;     // 0..3124
    const int m    = tm * 16 + mr;
    const int lowK = hf * 8;

    f8 c;
#pragma unroll
    for (int r = 0; r < 8; ++r)
        c[r] = out2[(tm * 16 + r + 8 * hf) * 16 + mr];

#pragma unroll
    for (int kk = 0; kk < 4; ++kk) {              // K = 128 = 4 x 32
        const float* ar = src + m * 128 + kk * 32;
        h16 a, b;
#pragma unroll
        for (int e = 0; e < 8; ++e) {
            a[e]     = (_Float16)ar[lowK + e];
            a[8 + e] = (_Float16)ar[16 + lowK + e];
        }
#pragma unroll
        for (int e = 0; e < 16; ++e)
            b[e] = W2sk[(kk * 32 + ((e >> 3) << 4) + lowK + (e & 7)) * 16 + mr];
        c = __builtin_amdgcn_wmma_f32_16x16x32_f16(false, a, false, b, (short)0, c,
                                                   false, false);
    }
#pragma unroll
    for (int r = 0; r < 8; ++r)
        out2[(tm * 16 + r + 8 * hf) * 16 + mr] = c[r];
}

// epilogue: mu = tanh(out2[:, :3]+b2[:3]); sigma = exp(2*tanh(out2[:, 3:6]+b2[3:6]))
__global__ void k_final(const float* __restrict__ out2, const float* __restrict__ b2,
                        float* __restrict__ out) {
    int idx = blockIdx.x * blockDim.x + threadIdx.x;   // 150000
    if (idx < N_NODES * 3) {
        int i = idx / 3, j = idx - i * 3;
        float mu = tanhf(out2[i * 16 + j] + b2[j]);
        float lv = tanhf(out2[i * 16 + 3 + j] + b2[3 + j]);
        out[idx] = mu;
        out[N_NODES * 3 + idx] = expf(2.f * lv);
    }
}

extern "C" void kernel_launch(void* const* d_in, const int* in_sizes, int n_in,
                              void* d_out, int out_size, void* d_ws, size_t ws_size,
                              hipStream_t stream) {
    (void)in_sizes; (void)n_in; (void)out_size; (void)ws_size;
    const float* x   = (const float*)d_in[0];
    const int*   ei  = (const int*)d_in[1];
    const int*   row = ei;
    const int*   col = ei + N_EDGES;
    const float* ew  = (const float*)d_in[2];
    const float* W1  = (const float*)d_in[3];
    const float* b1  = (const float*)d_in[4];
    const float* W2  = (const float*)d_in[5];
    const float* b2  = (const float*)d_in[6];
    float* out = (float*)d_out;

    // workspace layout (float elements, 64-aligned)
    float* ws   = (float*)d_ws;
    float* deg  = ws;                       // 50000  -> dis in place
    float* nw   = ws + 50048;               // 800000
    float* Xcat = ws + 850048;              // 50000*32
    float* hA   = ws + 2450048;             // 50000*128
    float* hB   = ws + 8850048;             // 50000*128
    float* out2 = ws + 15250048;            // 50000*16
    _Float16* W1s = (_Float16*)(ws + 16050048);     // 32*128 halfs
    _Float16* W2s = W1s + 32 * 128;                 // 5*128*16 halfs

    const int T = 256;
    // ---- normalization
    k_zero_f32<<<(N_NODES + T - 1) / T, T, 0, stream>>>(deg, N_NODES);
    k_deg<<<N_EDGES / T, T, 0, stream>>>(col, ew, deg);
    k_rsqrt<<<(N_NODES + T - 1) / T, T, 0, stream>>>(deg);
    k_normw<<<N_EDGES / T, T, 0, stream>>>(row, col, ew, deg, nw);

    // ---- weights + hop-0 features
    k_init_xcat<<<(N_NODES * 32) / T, T, 0, stream>>>(x, Xcat);
    k_prep_w1<<<(32 * 128) / T, T, 0, stream>>>(W1, W1s);
    k_prep_w2<<<(5 * 128 * 16) / T, T, 0, stream>>>(W2, W2s);

    // ---- layer 1: 4 hops into disjoint Xcat column bands, then one fused WMMA GEMM
    for (int k = 1; k <= 4; ++k)
        k_prop1<<<(N_EDGES * 8) / T, T, 0, stream>>>(row, col, nw, Xcat,
                                                     (k - 1) * 6, k * 6);
    k_gemm1<<<dim3(F_HID / 16, N_NODES / 16), 32, 0, stream>>>(Xcat, W1s, b1, hA);

    // ---- layer 2: hop -> WMMA GEMM accumulate into out2
    k_zero_f32<<<(N_NODES * 16) / T, T, 0, stream>>>(out2, N_NODES * 16);
    k_gemm2<<<N_NODES / 16, 32, 0, stream>>>(hA, W2s, out2);                  // k=0

    float* cur = hA, *nxt = hB;
    for (int k = 1; k <= 4; ++k) {
        k_zero_f32<<<(N_NODES * 128) / T, T, 0, stream>>>(nxt, N_NODES * 128);
        k_prop2<<<(N_EDGES * 128) / T, T, 0, stream>>>(row, col, nw, cur, nxt);
        k_gemm2<<<N_NODES / 16, 32, 0, stream>>>(nxt, W2s + k * 128 * 16, out2);
        float* t = cur; cur = nxt; nxt = t;
    }

    // ---- epilogue
    k_final<<<(N_NODES * 3 + T - 1) / T, T, 0, stream>>>(out2, b2, out);
}
